// SOCA_63960652972235
// MI455X (gfx1250) — matrix operand order repaired and checked by reading, not compiled, not run on hardware
//
#include <hip/hip_runtime.h>
#include <hip/hip_bf16.h>

typedef __attribute__((ext_vector_type(16))) __bf16 v16bf;
typedef __attribute__((ext_vector_type(8)))  float  v8f;
typedef __attribute__((ext_vector_type(2)))  float  v2f;

#define BATCH 16
#define CDIM  256
#define MDIM  9216          // 96*96
#define EPSV  1e-5f

// ---------------------------------------------------------------- helpers
__device__ inline __bf16 f2bf(float f) {
  unsigned u = __builtin_bit_cast(unsigned, f);
  u += 0x7FFFu + ((u >> 16) & 1u);            // round-to-nearest-even
  unsigned short h = (unsigned short)(u >> 16);
  return __builtin_bit_cast(__bf16, h);
}
__device__ inline float bf2f(__bf16 b) {
  unsigned short h = __builtin_bit_cast(unsigned short, b);
  unsigned u = ((unsigned)h) << 16;
  return __builtin_bit_cast(float, u);
}

union FragBF { v16bf v; uint4 q[2]; };

// Convert a float4 into hi/lo bf16 quads and store packed (8B) into LDS.
__device__ inline void cvt_store(__bf16* hi, __bf16* lo, int ofs, float4 v) {
  float f[4] = {v.x, v.y, v.z, v.w};
  unsigned short hs[4], ls[4];
#pragma unroll
  for (int e = 0; e < 4; ++e) {
    __bf16 hb = f2bf(f[e]);
    float  hf = bf2f(hb);
    __bf16 lb = f2bf(f[e] - hf);
    hs[e] = __builtin_bit_cast(unsigned short, hb);
    ls[e] = __builtin_bit_cast(unsigned short, lb);
  }
  uint2 hp = make_uint2((unsigned)hs[0] | ((unsigned)hs[1] << 16),
                        (unsigned)hs[2] | ((unsigned)hs[3] << 16));
  uint2 lp = make_uint2((unsigned)ls[0] | ((unsigned)ls[1] << 16),
                        (unsigned)ls[2] | ((unsigned)ls[3] << 16));
  *(uint2*)(hi + ofs) = hp;
  *(uint2*)(lo + ofs) = lp;
}

// ---------------------------------------------------------------- row means
__global__ __launch_bounds__(256) void row_mean_kernel(const float* __restrict__ x,
                                                       float* __restrict__ mean) {
  int row = blockIdx.x;                       // b*C + c
  const float* p = x + (size_t)row * MDIM;
  float s = 0.f;
  for (int i = threadIdx.x; i < MDIM; i += 256) s += p[i];
  __shared__ float red[256];
  red[threadIdx.x] = s; __syncthreads();
  for (int off = 128; off > 0; off >>= 1) {
    if (threadIdx.x < off) red[threadIdx.x] += red[threadIdx.x + off];
    __syncthreads();
  }
  if (threadIdx.x == 0) mean[row] = red[0] * (1.0f / (float)MDIM);
}

// ---------------------------------------------------------------- Gram partials (bf16x3 WMMA)
#define GT     128
#define GK     32
#define GLD    40                 // bf16 row stride (16B-aligned frags + bank spread)
#define SPLITK 4
#define GSTEPS (MDIM / (SPLITK * GK))   // 72
#define GBUFE  (4 * GT * GLD)           // elements per LDS buffer (Ahi,Alo,Bhi,Blo)
#define GRAM_SMEM (2 * GBUFE * 2)       // bytes: double buffered

__global__ __launch_bounds__(256) void gram_cov_kernel(const float* __restrict__ x,
                                                       float* __restrict__ covP) {
  const int b  = blockIdx.z;
  const int ty = blockIdx.y;
  const int tx = blockIdx.x & 1;
  const int sp = blockIdx.x >> 1;          // K split index
  const float* X = x + (size_t)b * CDIM * MDIM;
  const int k0 = sp * (MDIM / SPLITK);

  extern __shared__ char dynsm[];
  __bf16* sm = (__bf16*)dynsm;

  const int tid  = threadIdx.x;
  const int lane = tid & 31;
  const int wid  = tid >> 5;     // 8 waves
  const int wm   = wid & 1;      // 2 row blocks of 64
  const int wn   = wid >> 1;     // 4 col blocks of 32
  const int lm   = lane & 15;
  const int lh   = lane >> 4;

  const int row0 = ty * GT;
  const int col0 = tx * GT;

  // per-thread staging coordinates: 4 float4 per tile
  int fr[4], fc[4];
#pragma unroll
  for (int j = 0; j < 4; ++j) {
    int f = tid + j * 256;
    fr[j] = f >> 3;
    fc[j] = (f & 7) * 4;
  }

  v8f acc[4][2];
#pragma unroll
  for (int i = 0; i < 4; ++i)
#pragma unroll
    for (int j = 0; j < 2; ++j) acc[i][j] = (v8f)0.0f;

  // prologue: stage tile 0 into buffer 0
  {
    __bf16* Ahi = sm;            __bf16* Alo = sm + GT * GLD;
    __bf16* Bhi = sm + 2 * GT * GLD; __bf16* Blo = sm + 3 * GT * GLD;
#pragma unroll
    for (int j = 0; j < 4; ++j) {
      float4 av = *(const float4*)(X + (size_t)(row0 + fr[j]) * MDIM + k0 + fc[j]);
      float4 bv = *(const float4*)(X + (size_t)(col0 + fr[j]) * MDIM + k0 + fc[j]);
      cvt_store(Ahi, Alo, fr[j] * GLD + fc[j], av);
      cvt_store(Bhi, Blo, fr[j] * GLD + fc[j], bv);
    }
  }
  __syncthreads();

  for (int step = 0; step < GSTEPS; ++step) {
    const int cur = step & 1;
    const __bf16* Ahi = sm + cur * GBUFE;
    const __bf16* Alo = Ahi + GT * GLD;
    const __bf16* Bhi = Alo + GT * GLD;
    const __bf16* Blo = Bhi + GT * GLD;

    // issue next tile's global loads (latency hidden behind WMMA below)
    const bool more = (step + 1) < GSTEPS;
    float4 na[4], nb[4];
    if (more) {
      int kk = k0 + (step + 1) * GK;
#pragma unroll
      for (int j = 0; j < 4; ++j) {
        na[j] = *(const float4*)(X + (size_t)(row0 + fr[j]) * MDIM + kk + fc[j]);
        nb[j] = *(const float4*)(X + (size_t)(col0 + fr[j]) * MDIM + kk + fc[j]);
      }
    }
    // deep prefetch into GL2 (global_prefetch_b8)
    if (step + 4 < GSTEPS) {
      int kp = k0 + (step + 4) * GK;
#pragma unroll
      for (int j = 0; j < 4; ++j) {
        __builtin_prefetch(X + (size_t)(row0 + fr[j]) * MDIM + kp + fc[j], 0, 3);
        __builtin_prefetch(X + (size_t)(col0 + fr[j]) * MDIM + kp + fc[j], 0, 3);
      }
    }

    // fragments + 24 WMMA (hi*hi + hi*lo + lo*hi)
    FragBF ah[4], al[4], bh[2], bl[2];
#pragma unroll
    for (int mt = 0; mt < 4; ++mt) {
      int r = wm * 64 + mt * 16 + lm;
      int kofs = lh * 8;
      ah[mt].q[0] = *(const uint4*)&Ahi[r * GLD + kofs];
      ah[mt].q[1] = *(const uint4*)&Ahi[r * GLD + kofs + 16];
      al[mt].q[0] = *(const uint4*)&Alo[r * GLD + kofs];
      al[mt].q[1] = *(const uint4*)&Alo[r * GLD + kofs + 16];
    }
#pragma unroll
    for (int nt = 0; nt < 2; ++nt) {
      int n  = wn * 32 + nt * 16 + lm;
      int kb = lh * 8;
      bh[nt].q[0] = *(const uint4*)&Bhi[n * GLD + kb];
      bh[nt].q[1] = *(const uint4*)&Bhi[n * GLD + kb + 16];
      bl[nt].q[0] = *(const uint4*)&Blo[n * GLD + kb];
      bl[nt].q[1] = *(const uint4*)&Blo[n * GLD + kb + 16];
    }
#pragma unroll
    for (int mt = 0; mt < 4; ++mt)
#pragma unroll
      for (int nt = 0; nt < 2; ++nt) {
        v8f c = acc[mt][nt];
        c = __builtin_amdgcn_wmma_f32_16x16x32_bf16(false, ah[mt].v, false, bh[nt].v, (short)0, c, false, false);
        c = __builtin_amdgcn_wmma_f32_16x16x32_bf16(false, ah[mt].v, false, bl[nt].v, (short)0, c, false, false);
        c = __builtin_amdgcn_wmma_f32_16x16x32_bf16(false, al[mt].v, false, bh[nt].v, (short)0, c, false, false);
        acc[mt][nt] = c;
      }

    __syncthreads();                 // everyone done reading buffer `cur`
    if (more) {
      __bf16* wA = sm + (cur ^ 1) * GBUFE;
#pragma unroll
      for (int j = 0; j < 4; ++j) {
        cvt_store(wA,                wA + GT * GLD,     fr[j] * GLD + fc[j], na[j]);
        cvt_store(wA + 2 * GT * GLD, wA + 3 * GT * GLD, fr[j] * GLD + fc[j], nb[j]);
      }
    }
    __syncthreads();                 // buffer `cur^1` ready for next step
  }

  // raw partial sums (mean/1/M folded in by combine kernel)
  float* cb = covP + ((size_t)sp * BATCH + b) * CDIM * CDIM;
#pragma unroll
  for (int mt = 0; mt < 4; ++mt)
#pragma unroll
    for (int nt = 0; nt < 2; ++nt) {
      int col   = col0 + wn * 32 + nt * 16 + lm;
      int rbase = row0 + wm * 64 + mt * 16 + lh * 8;
#pragma unroll
      for (int vi = 0; vi < 8; ++vi)
        cb[(rbase + vi) * CDIM + col] = acc[mt][nt][vi];
    }
}

// ---------------------------------------------------------------- combine splits + epilogue
__global__ __launch_bounds__(256) void combine_cov_kernel(const float* __restrict__ covP,
                                                          const float* __restrict__ mean,
                                                          float* __restrict__ cov) {
  int idx = blockIdx.x * 256 + threadIdx.x;
  if (idx >= BATCH * CDIM * CDIM) return;
  int b = idx >> 16;
  int rc = idx & 65535;
  int r = rc >> 8, c = rc & 255;
  float s = 0.f;
#pragma unroll
  for (int k = 0; k < SPLITK; ++k)
    s += covP[(size_t)k * BATCH * CDIM * CDIM + idx];
  const float* mb = mean + b * CDIM;
  cov[idx] = s * (1.0f / (float)MDIM) - mb[r] * mb[c];
}

// ---------------------------------------------------------------- normA = sum(cov[b])
__global__ __launch_bounds__(256) void norm_kernel(const float* __restrict__ cov,
                                                   float* __restrict__ normA) {
  int b = blockIdx.x;
  const float* p = cov + (size_t)b * CDIM * CDIM;
  float s = 0.f;
  for (int i = threadIdx.x; i < CDIM * CDIM; i += 256) s += p[i];
  __shared__ float red[256];
  red[threadIdx.x] = s; __syncthreads();
  for (int off = 128; off > 0; off >>= 1) {
    if (threadIdx.x < off) red[threadIdx.x] += red[threadIdx.x + off];
    __syncthreads();
  }
  if (threadIdx.x == 0) normA[b] = red[0];
}

// ---------------------------------------------------------------- A = cov/(n+eps); Z0 = 1.5I - 0.5A
__global__ __launch_bounds__(256) void prep_kernel(const float* __restrict__ cov,
                                                   const float* __restrict__ normA,
                                                   float* __restrict__ A,
                                                   float* __restrict__ Z) {
  int idx = blockIdx.x * 256 + threadIdx.x;
  if (idx >= BATCH * CDIM * CDIM) return;
  int b  = idx >> 16;
  int rc = idx & 65535;
  int r = rc >> 8, c = rc & 255;
  float a = cov[idx] / (normA[b] + EPSV);
  A[idx] = a;
  Z[idx] = (r == c ? 1.5f : 0.0f) - 0.5f * a;
}

// ---------------------------------------------------------------- Newton-Schulz GEMM (fp32 WMMA)
// D = c0*I + c1*(A@B) + c2*E   (all [B,256,256] fp32), double-buffered LDS
#define ALD 36
#define BLD 33
#define NSBUF (128 * ALD + 128 * BLD)       // floats per buffer (At + Bt)
#define NS_SMEM (2 * NSBUF * 4)             // bytes
#define NSTEP (CDIM / 32)                   // 8

__global__ __launch_bounds__(256) void ns_gemm_kernel(const float* __restrict__ A,
                                                      const float* __restrict__ Bm,
                                                      const float* __restrict__ E,
                                                      float* __restrict__ D,
                                                      float c0, float c1, float c2) {
  const int b  = blockIdx.z;
  const int ty = blockIdx.y;
  const int tx = blockIdx.x;
  const float* Ab = A  + (size_t)b * CDIM * CDIM;
  const float* Bb = Bm + (size_t)b * CDIM * CDIM;

  extern __shared__ char dynsm[];
  float* fsm = (float*)dynsm;

  const int tid  = threadIdx.x;
  const int lane = tid & 31;
  const int wid  = tid >> 5;
  const int wm   = wid & 1;
  const int wn   = wid >> 1;
  const int lm   = lane & 15;
  const int lh   = lane >> 4;

  // staging coords: A tile 128x32 (1024 float4), B tile 32x128 (1024 float4)
  int ar[4], ac[4], bk[4], bc[4];
#pragma unroll
  for (int j = 0; j < 4; ++j) {
    int f = tid + j * 256;
    ar[j] = f >> 3;  ac[j] = (f & 7) * 4;
    bk[j] = f >> 5;  bc[j] = (f & 31) * 4;
  }

  v8f acc[4][2];
#pragma unroll
  for (int i = 0; i < 4; ++i)
#pragma unroll
    for (int j = 0; j < 2; ++j) acc[i][j] = (v8f)0.0f;

  const int row0 = ty * 128;
  const int col0 = tx * 128;

  // prologue: stage step 0 into buffer 0
  {
    float* At = fsm;
    float* Bt = fsm + 128 * ALD;
#pragma unroll
    for (int j = 0; j < 4; ++j) {
      float4 av = *(const float4*)(Ab + (size_t)(row0 + ar[j]) * CDIM + ac[j]);
      *(float4*)&At[ar[j] * ALD + ac[j]] = av;
      float4 bv = *(const float4*)(Bb + (size_t)bk[j] * CDIM + col0 + bc[j]);
      Bt[(bc[j] + 0) * BLD + bk[j]] = bv.x;
      Bt[(bc[j] + 1) * BLD + bk[j]] = bv.y;
      Bt[(bc[j] + 2) * BLD + bk[j]] = bv.z;
      Bt[(bc[j] + 3) * BLD + bk[j]] = bv.w;
    }
  }
  __syncthreads();

  for (int step = 0; step < NSTEP; ++step) {
    const int cur = step & 1;
    const float* At = fsm + cur * NSBUF;
    const float* Bt = At + 128 * ALD;

    const bool more = (step + 1) < NSTEP;
    float4 na[4], nb[4];
    if (more) {
      int kk = (step + 1) * 32;
#pragma unroll
      for (int j = 0; j < 4; ++j) {
        na[j] = *(const float4*)(Ab + (size_t)(row0 + ar[j]) * CDIM + kk + ac[j]);
        nb[j] = *(const float4*)(Bb + (size_t)(kk + bk[j]) * CDIM + col0 + bc[j]);
      }
    }

#pragma unroll
    for (int k4 = 0; k4 < 32; k4 += 4) {
      v2f a[4], bb[2];
#pragma unroll
      for (int mt = 0; mt < 4; ++mt) {
        int r = wm * 64 + mt * 16 + lm;
        a[mt] = *(const v2f*)&At[r * ALD + k4 + 2 * lh];
      }
#pragma unroll
      for (int nt = 0; nt < 2; ++nt) {
        int n = wn * 32 + nt * 16 + lm;
        v2f t;
        t[0] = Bt[n * BLD + k4 + 2 * lh];
        t[1] = Bt[n * BLD + k4 + 2 * lh + 1];
        bb[nt] = t;
      }
#pragma unroll
      for (int mt = 0; mt < 4; ++mt)
#pragma unroll
        for (int nt = 0; nt < 2; ++nt)
          acc[mt][nt] = __builtin_amdgcn_wmma_f32_16x16x4_f32(
              false, a[mt], false, bb[nt], (short)0, acc[mt][nt], false, false);
    }

    __syncthreads();
    if (more) {
      float* wAt = fsm + (cur ^ 1) * NSBUF;
      float* wBt = wAt + 128 * ALD;
#pragma unroll
      for (int j = 0; j < 4; ++j) {
        *(float4*)&wAt[ar[j] * ALD + ac[j]] = na[j];
        wBt[(bc[j] + 0) * BLD + bk[j]] = nb[j].x;
        wBt[(bc[j] + 1) * BLD + bk[j]] = nb[j].y;
        wBt[(bc[j] + 2) * BLD + bk[j]] = nb[j].z;
        wBt[(bc[j] + 3) * BLD + bk[j]] = nb[j].w;
      }
    }
    __syncthreads();
  }

  float* Db = D + (size_t)b * CDIM * CDIM;
  const float* Eb = E ? (E + (size_t)b * CDIM * CDIM) : nullptr;
#pragma unroll
  for (int mt = 0; mt < 4; ++mt)
#pragma unroll
    for (int nt = 0; nt < 2; ++nt) {
      int col   = col0 + wn * 32 + nt * 16 + lm;
      int rbase = row0 + wm * 64 + mt * 16 + lh * 8;
#pragma unroll
      for (int vi = 0; vi < 8; ++vi) {
        int row = rbase + vi;
        float v = c1 * acc[mt][nt][vi];
        if (row == col) v += c0;
        if (Eb) v += c2 * Eb[row * CDIM + col];
        Db[row * CDIM + col] = v;
      }
    }
}

// ---------------------------------------------------------------- column mean * sqrt(normA+eps)
__global__ __launch_bounds__(256) void colmean_kernel(const float* __restrict__ T,
                                                      const float* __restrict__ normA,
                                                      float* __restrict__ yv) {
  int b = blockIdx.x, c = threadIdx.x;
  const float* Tb = T + (size_t)b * CDIM * CDIM;
  float s = 0.f;
  for (int i = 0; i < CDIM; ++i) s += Tb[i * CDIM + c];
  yv[b * CDIM + c] = (s * (1.0f / (float)CDIM)) * sqrtf(normA[b] + EPSV);
}

// ---------------------------------------------------------------- center-tap "convs" (relu, sigmoid)
__global__ __launch_bounds__(256) void mix_kernel(const float* __restrict__ yv,
                                                  const float* __restrict__ w1,
                                                  const float* __restrict__ b1,
                                                  const float* __restrict__ w2,
                                                  const float* __restrict__ b2,
                                                  float* __restrict__ g) {
  int b = blockIdx.x, o = threadIdx.x;
  __shared__ float yin[CDIM], g1[CDIM];
  yin[o] = yv[b * CDIM + o];
  __syncthreads();
  float s = b1[o];
  for (int i = 0; i < CDIM; ++i) s += w1[(size_t)(o * CDIM + i) * 9 + 4] * yin[i];
  g1[o] = fmaxf(s, 0.0f);
  __syncthreads();
  float t = b2[o];
  for (int i = 0; i < CDIM; ++i) t += w2[(size_t)(o * CDIM + i) * 9 + 4] * g1[i];
  g[b * CDIM + o] = 1.0f / (1.0f + __expf(-t));
}

// ---------------------------------------------------------------- out = g[b,c] * x
__global__ __launch_bounds__(256) void scale_kernel(const float* __restrict__ x,
                                                    const float* __restrict__ g,
                                                    float* __restrict__ out,
                                                    long long n4) {
  long long i = (long long)blockIdx.x * 256 + threadIdx.x;
  if (i >= n4) return;
  long long base = i * 4;
  int ch = (int)(base / MDIM);                 // = b*C + c (M divisible by 4)
  float gg = g[ch];
  float4 v = ((const float4*)x)[i];
  v.x *= gg; v.y *= gg; v.z *= gg; v.w *= gg;
  ((float4*)out)[i] = v;
}

// ---------------------------------------------------------------- host
extern "C" void kernel_launch(void* const* d_in, const int* in_sizes, int n_in,
                              void* d_out, int out_size, void* d_ws, size_t ws_size,
                              hipStream_t stream) {
  const float* x  = (const float*)d_in[0];
  const float* w1 = (const float*)d_in[1];
  const float* b1 = (const float*)d_in[2];
  const float* w2 = (const float*)d_in[3];
  const float* b2 = (const float*)d_in[4];
  float* out = (float*)d_out;

  float* wsf   = (float*)d_ws;
  const size_t NM = (size_t)BATCH * CDIM * CDIM;   // 1,048,576 floats per matrix set
  float* mean  = wsf;               // 4096
  float* normA = wsf + 4096;        // 16
  float* yv    = wsf + 4224;        // 4096
  float* gv    = wsf + 8320;        // 4096
  float* cov   = wsf + 16384;
  float* covP  = cov  + NM;         // SPLITK partial grams
  float* bA    = covP + (size_t)SPLITK * NM;
  float* bY    = bA  + NM;
  float* bZ    = bY  + NM;
  float* bT    = bZ  + NM;
  float* bU    = bT  + NM;
  float* bV    = bU  + NM;

  row_mean_kernel<<<BATCH * CDIM, 256, 0, stream>>>(x, mean);

  gram_cov_kernel<<<dim3(2 * SPLITK, 2, BATCH), 256, GRAM_SMEM, stream>>>(x, covP);
  combine_cov_kernel<<<(int)((BATCH * CDIM * CDIM + 255) / 256), 256, 0, stream>>>(covP, mean, cov);

  norm_kernel<<<BATCH, 256, 0, stream>>>(cov, normA);
  prep_kernel<<<(int)((BATCH * CDIM * CDIM + 255) / 256), 256, 0, stream>>>(cov, normA, bA, bZ);

  dim3 gg(2, 2, BATCH);
  // Y = A @ ZY0 ; Z = ZY0
  ns_gemm_kernel<<<gg, 256, NS_SMEM, stream>>>(bA, bZ, nullptr, bY, 0.f, 1.f, 0.f);

  float* Y = bY; float* Z = bZ; float* U = bU; float* V = bV;
  for (int it = 0; it < 3; ++it) {
    ns_gemm_kernel<<<gg, 256, NS_SMEM, stream>>>(Z, Y, nullptr, bT, 1.5f, -0.5f, 0.f); // ZY
    ns_gemm_kernel<<<gg, 256, NS_SMEM, stream>>>(Y, bT, nullptr, U, 0.f, 1.f, 0.f);    // Y'
    ns_gemm_kernel<<<gg, 256, NS_SMEM, stream>>>(bT, Z, nullptr, V, 0.f, 1.f, 0.f);    // Z'
    float* t;
    t = Y; Y = U; U = t;
    t = Z; Z = V; V = t;
  }
  // final: T = Z@Y ; R = 1.5*Y - 0.5*(Y@T)
  ns_gemm_kernel<<<gg, 256, NS_SMEM, stream>>>(Z, Y, nullptr, bT, 0.f, 1.f, 0.f);
  ns_gemm_kernel<<<gg, 256, NS_SMEM, stream>>>(Y, bT, Y, U, 0.f, -0.5f, 1.5f);

  colmean_kernel<<<BATCH, 256, 0, stream>>>(U, normA, yv);
  mix_kernel<<<BATCH, 256, 0, stream>>>(yv, w1, b1, w2, b2, gv);

  long long n4 = (long long)BATCH * CDIM * MDIM / 4;
  scale_kernel<<<(unsigned)((n4 + 255) / 256), 256, 0, stream>>>(x, gv, out, n4);
}
